// RelationGCN_38637525795190
// MI455X (gfx1250) — matrix operand order repaired and checked by reading, not compiled, not run on hardware
//
#include <hip/hip_runtime.h>

#define NNODES 50000
#define NEDGES 800000
#define DIM    128
#define NDTOT  (NNODES * DIM)

typedef __attribute__((ext_vector_type(16))) __bf16 v16bf;
typedef __attribute__((ext_vector_type(8)))  __bf16 v8bf;
typedef __attribute__((ext_vector_type(8)))  float  v8f;

// ---------------- utility kernels ----------------
__global__ void k_zero(float* __restrict__ p, int n) {
    int i = blockIdx.x * blockDim.x + threadIdx.x;
    if (i < n) p[i] = 0.0f;
}

__global__ void k_copy4(const float* __restrict__ s, float* __restrict__ d0,
                        float* __restrict__ d1, float* __restrict__ d2,
                        float* __restrict__ d3, int n) {
    int i = blockIdx.x * blockDim.x + threadIdx.x;
    if (i < n) { float v = s[i]; d0[i] = v; d1[i] = v; d2[i] = v; d3[i] = v; }
}

__global__ void k_copy(const float* __restrict__ s, float* __restrict__ d, int n) {
    int i = blockIdx.x * blockDim.x + threadIdx.x;
    if (i < n) d[i] = s[i];
}

// ---------------- degree / normalizers ----------------
__global__ void k_deg(const int* __restrict__ ei, float* __restrict__ deg_out,
                      float* __restrict__ deg_in) {
    int i = blockIdx.x * blockDim.x + threadIdx.x;
    if (i < NEDGES) {
        atomicAdd(&deg_out[ei[i]], 1.0f);
        atomicAdd(&deg_in[ei[NEDGES + i]], 1.0f);
    }
}

__global__ void k_invsqrt(float* __restrict__ p, int n) {
    int i = blockIdx.x * blockDim.x + threadIdx.x;
    if (i < n) { float d = p[i]; p[i] = (d > 0.0f) ? rsqrtf(d) : 1.0f; }
}

// ---------------- GEMM prep ----------------
// A_bf[row][k] = bf16( emb[row][k] * rel[k] * c_src[row] )
__global__ void k_prepA(const float* __restrict__ emb, const float* __restrict__ rel,
                        const float* __restrict__ csrc, __bf16* __restrict__ A) {
    int i = blockIdx.x * blockDim.x + threadIdx.x;
    if (i < NDTOT) {
        int row = i >> 7, col = i & 127;
        A[i] = (__bf16)(emb[i] * rel[col] * csrc[row]);
    }
}

// Wt[n][k] = bf16( W[k][n] )  (column-contiguous-in-k for WMMA B operand loads)
__global__ void k_packW(const float* __restrict__ W, __bf16* __restrict__ Wt) {
    int i = blockIdx.x * blockDim.x + threadIdx.x;
    if (i < DIM * DIM) {
        int k = i >> 7, n = i & 127;
        Wt[n * DIM + k] = (__bf16)W[i];
    }
}

// ---------------- WMMA GEMM: H[N,128] = A[N,128] @ W[128,128] ----------------
// block = 256 threads (8 waves); block -> one 16-row M tile; wave w -> N tile w.
__global__ void __launch_bounds__(256)
k_gemm(const __bf16* __restrict__ A, const __bf16* __restrict__ Wt,
       float* __restrict__ H) {
    const int m0   = blockIdx.x * 16;               // 3125 blocks, exact
    const int wave = threadIdx.x >> 5;
    const int lane = threadIdx.x & 31;
    const int n0   = wave * 16;
    const int r    = lane & 15;
    const int half = lane >> 4;
    const int arow = m0 + r;                        // A row this lane supplies
    const int bcol = n0 + r;                        // B column this lane supplies

    v8f acc = {};
#pragma unroll
    for (int ks = 0; ks < 4; ++ks) {                // K = 128 in 4 steps of 32
        const int kb = ks * 32 + half * 8;
        v8bf alo = *(const v8bf*)(A + (size_t)arow * DIM + kb);
        v8bf ahi = *(const v8bf*)(A + (size_t)arow * DIM + kb + 16);
        v16bf a  = __builtin_shufflevector(alo, ahi,
                     0,1,2,3,4,5,6,7,8,9,10,11,12,13,14,15);
        v8bf blo = *(const v8bf*)(Wt + (size_t)bcol * DIM + kb);
        v8bf bhi = *(const v8bf*)(Wt + (size_t)bcol * DIM + kb + 16);
        v16bf b  = __builtin_shufflevector(blo, bhi,
                     0,1,2,3,4,5,6,7,8,9,10,11,12,13,14,15);
        acc = __builtin_amdgcn_wmma_f32_16x16x32_bf16(
                  false, a, false, b, (short)0, acc, false, false);
    }
#pragma unroll
    for (int rr = 0; rr < 8; ++rr) {
        // C/D layout: VGPR rr, lanes0-15 -> M=rr, lanes16-31 -> M=rr+8
        H[(size_t)(m0 + rr + half * 8) * DIM + n0 + r] = acc[rr];
    }
}

// ---------------- edge gather + scatter-add ----------------
// 32 threads per edge; thread covers 4 features; prefetch next gather row.
__global__ void __launch_bounds__(256)
k_scatter(const float* __restrict__ H, const int* __restrict__ ei,
          float* __restrict__ agg) {
    const long long t = (long long)blockIdx.x * blockDim.x + threadIdx.x;
    const int edge = (int)(t >> 5);
    if (edge >= NEDGES) return;
    const int c4 = ((int)t & 31) * 4;
    const int s = ei[edge];
    const int d = ei[NEDGES + edge];
    if (edge + 8 < NEDGES) {
        int sn = ei[edge + 8];
        __builtin_prefetch(H + (size_t)sn * DIM + c4, 0, 0);   // global_prefetch_b8
    }
    const float4 v = *(const float4*)(H + (size_t)s * DIM + c4);
    float* a = agg + (size_t)d * DIM + c4;
    atomicAdd(a + 0, v.x);
    atomicAdd(a + 1, v.y);
    atomicAdd(a + 2, v.z);
    atomicAdd(a + 3, v.w);
}

// ---------------- conv epilogue + BN stats ----------------
// v = agg * c_dst + b ; overwrite H with v ; accumulate per-column sum/sumsq
__global__ void __launch_bounds__(128)
k_finalize_stats(const float* __restrict__ agg, const float* __restrict__ cdst,
                 const float* __restrict__ bias, float* __restrict__ H,
                 float* __restrict__ stats) {
    const int col = threadIdx.x;
    const float b = bias[col];
    float sum = 0.0f, sq = 0.0f;
    for (int row = blockIdx.x; row < NNODES; row += gridDim.x) {
        float v = agg[(size_t)row * DIM + col] * cdst[row] + b;
        H[(size_t)row * DIM + col] = v;
        sum += v; sq += v * v;
    }
    atomicAdd(&stats[col], sum);
    atomicAdd(&stats[DIM + col], sq);
}

__global__ void k_bnparams(const float* __restrict__ stats,
                           const float* __restrict__ gamma,
                           const float* __restrict__ beta,
                           float* __restrict__ bnp) {
    int c = threadIdx.x;
    const float inv_n = 1.0f / (float)NNODES;
    float mu  = stats[c] * inv_n;
    float var = stats[DIM + c] * inv_n - mu * mu;
    float sc  = rsqrtf(var + 1e-5f) * gamma[c];
    bnp[c]       = sc;
    bnp[DIM + c] = beta[c] - mu * sc;
}

// emb += leaky_relu( H*scale + shift )
__global__ void k_update_emb(const float* __restrict__ H, const float* __restrict__ bnp,
                             float* __restrict__ emb) {
    int i = blockIdx.x * blockDim.x + threadIdx.x;
    if (i < NDTOT) {
        int c = i & 127;
        float hn = H[i] * bnp[c] + bnp[DIM + c];
        emb[i] += (hn > 0.0f) ? hn : 0.01f * hn;
    }
}

// final layer: out = agg * c_dst + b (no BN / residual)
__global__ void k_final_out(const float* __restrict__ agg, const float* __restrict__ cdst,
                            const float* __restrict__ bias, float* __restrict__ out) {
    int i = blockIdx.x * blockDim.x + threadIdx.x;
    if (i < NDTOT) {
        int row = i >> 7, c = i & 127;
        out[i] = agg[i] * cdst[row] + bias[c];
    }
}

// out[j] = rel_b[j] + sum_k rel[k] * rel_W[j][k]
__global__ void __launch_bounds__(128)
k_rel_update(const float* __restrict__ rel, const float* __restrict__ rW,
             const float* __restrict__ rb, float* __restrict__ out) {
    __shared__ float s[DIM];
    int j = threadIdx.x;
    s[j] = rel[j];
    __syncthreads();
    float acc = rb[j];
#pragma unroll 4
    for (int k = 0; k < DIM; ++k) acc += s[k] * rW[j * DIM + k];
    out[j] = acc;
}

// ---------------- host orchestration ----------------
extern "C" void kernel_launch(void* const* d_in, const int* in_sizes, int n_in,
                              void* d_out, int out_size, void* d_ws, size_t ws_size,
                              hipStream_t stream) {
    const float* features = (const float*)d_in[0];
    const float* r_in[4]  = {(const float*)d_in[1], (const float*)d_in[2],
                             (const float*)d_in[3], (const float*)d_in[4]};
    const int*   ei[4]    = {(const int*)d_in[5], (const int*)d_in[6],
                             (const int*)d_in[7], (const int*)d_in[8]};
    const float* gcn_W    = (const float*)d_in[9];
    const float* gcn_b    = (const float*)d_in[10];
    const float* bn_gamma = (const float*)d_in[11];
    const float* bn_beta  = (const float*)d_in[12];
    const float* rel_W    = (const float*)d_in[13];
    const float* rel_b    = (const float*)d_in[14];
    float* out = (float*)d_out;

    // workspace carve-out (256B aligned slabs)
    char* wp = (char*)d_ws;
    auto carve = [&](size_t bytes) -> char* {
        char* p = wp; wp += (bytes + 255) & ~(size_t)255; return p;
    };
    float*  c_src  = (float*)carve(4 * NNODES * 4);
    float*  c_dst  = (float*)carve(4 * NNODES * 4);
    float*  embs[4];
    for (int t = 0; t < 4; ++t) embs[t] = (float*)carve((size_t)NDTOT * 4);
    float*  h      = (float*)carve((size_t)NDTOT * 4);
    float*  agg    = (float*)carve((size_t)NDTOT * 4);
    __bf16* Abf    = (__bf16*)carve((size_t)NDTOT * 2);
    __bf16* Wt     = (__bf16*)carve(DIM * DIM * 2);
    float*  relbuf = (float*)carve(4 * 2 * DIM * 4);   // [rel][pingpong][DIM]
    float*  stats  = (float*)carve(2 * DIM * 4);
    float*  bnp    = (float*)carve(2 * DIM * 4);

    const int BLK = 256;
    const int g_nd   = (NDTOT + BLK - 1) / BLK;          // 25000
    const int g_edge = (NEDGES + BLK - 1) / BLK;         // 3125
    const int g_scat = (int)(((long long)NEDGES * 32 + BLK - 1) / BLK);  // 100000
    const int g_gemm = NNODES / 16;                      // 3125 (exact)

    // --- degrees + normalizers (edges constant across layers) ---
    k_zero<<<(4 * NNODES + BLK - 1) / BLK, BLK, 0, stream>>>(c_src, 4 * NNODES);
    k_zero<<<(4 * NNODES + BLK - 1) / BLK, BLK, 0, stream>>>(c_dst, 4 * NNODES);
    for (int t = 0; t < 4; ++t)
        k_deg<<<g_edge, BLK, 0, stream>>>(ei[t], c_src + t * NNODES, c_dst + t * NNODES);
    k_invsqrt<<<(4 * NNODES + BLK - 1) / BLK, BLK, 0, stream>>>(c_src, 4 * NNODES);
    k_invsqrt<<<(4 * NNODES + BLK - 1) / BLK, BLK, 0, stream>>>(c_dst, 4 * NNODES);

    // --- init state ---
    k_copy4<<<g_nd, BLK, 0, stream>>>(features, embs[0], embs[1], embs[2], embs[3], NDTOT);
    for (int t = 0; t < 4; ++t)
        k_copy<<<1, DIM, 0, stream>>>(r_in[t], relbuf + (t * 2 + 0) * DIM, DIM);

    int cur = 0;
    // --- layers 0..1 : conv + BN + leaky residual + rel update ---
    for (int layer = 0; layer < 2; ++layer) {
        k_packW<<<(DIM * DIM + BLK - 1) / BLK, BLK, 0, stream>>>(gcn_W + layer * DIM * DIM, Wt);
        for (int t = 0; t < 4; ++t) {
            const float* rel = relbuf + (t * 2 + cur) * DIM;
            k_prepA<<<g_nd, BLK, 0, stream>>>(embs[t], rel, c_src + t * NNODES, Abf);
            k_gemm<<<g_gemm, 256, 0, stream>>>(Abf, Wt, h);
            k_zero<<<g_nd, BLK, 0, stream>>>(agg, NDTOT);
            k_zero<<<1, 256, 0, stream>>>(stats, 2 * DIM);
            k_scatter<<<g_scat, BLK, 0, stream>>>(h, ei[t], agg);
            k_finalize_stats<<<256, DIM, 0, stream>>>(agg, c_dst + t * NNODES,
                                                      gcn_b + layer * DIM, h, stats);
            k_bnparams<<<1, DIM, 0, stream>>>(stats, bn_gamma + layer * DIM,
                                              bn_beta + layer * DIM, bnp);
            k_update_emb<<<g_nd, BLK, 0, stream>>>(h, bnp, embs[t]);
            k_rel_update<<<1, DIM, 0, stream>>>(rel, rel_W + layer * DIM * DIM,
                                                rel_b + layer * DIM,
                                                relbuf + (t * 2 + (cur ^ 1)) * DIM);
        }
        cur ^= 1;
    }

    // --- final layer: conv only + rel update, straight to outputs ---
    k_packW<<<(DIM * DIM + BLK - 1) / BLK, BLK, 0, stream>>>(gcn_W + 2 * DIM * DIM, Wt);
    for (int t = 0; t < 4; ++t) {
        const float* rel = relbuf + (t * 2 + cur) * DIM;
        k_prepA<<<g_nd, BLK, 0, stream>>>(embs[t], rel, c_src + t * NNODES, Abf);
        k_gemm<<<g_gemm, 256, 0, stream>>>(Abf, Wt, h);
        k_zero<<<g_nd, BLK, 0, stream>>>(agg, NDTOT);
        k_scatter<<<g_scat, BLK, 0, stream>>>(h, ei[t], agg);
        k_final_out<<<g_nd, BLK, 0, stream>>>(agg, c_dst + t * NNODES,
                                              gcn_b + 2 * DIM,
                                              out + (size_t)t * NDTOT);
        k_rel_update<<<1, DIM, 0, stream>>>(rel, rel_W + 2 * DIM * DIM,
                                            rel_b + 2 * DIM,
                                            out + (size_t)4 * NDTOT + t * DIM);
    }
    (void)in_sizes; (void)n_in; (void)out_size; (void)ws_size;
}
